// GNN_24816321036584
// MI455X (gfx1250) — matrix operand order repaired and checked by reading, not compiled, not run on hardware
//
#include <hip/hip_runtime.h>
#include <hip/hip_bf16.h>
#include <math.h>
#include <stdint.h>

// Problem constants (from reference)
#define U_   32
#define SU_  8
#define H_   64
#define NC_  4
#define B_   1024
#define E_   992           // U*(U-1); 992 = 62 * 16 exactly
#define NT_  62            // number of 16-edge M tiles
#define NITER 3
#define STRIDE 72          // staging row stride in halves (64 + 8 pad)
#define G3H  192           // 3*H gate width

typedef __attribute__((ext_vector_type(16))) _Float16 v16h;
typedef __attribute__((ext_vector_type(2)))  _Float16 v2h;
typedef __attribute__((ext_vector_type(2)))  __fp16   v2hf;   // cvt_pkrtz result type
typedef __attribute__((ext_vector_type(8)))  float    v8f;

// Compiler-only ordering fence: same-wave LDS ops are processed in order by
// hardware (ISA 7.3), we only need to stop compile-time reordering.
#define LDS_ORDER() asm volatile("" ::: "memory")

// ---- CDNA5 WMMA 16x16x32 f16 register layouts (CDNA5 ISA) ----
// A (16x32, MxK): lane<16 -> row M=lane, K = {0..7, 16..23};
//                 lane>=16 -> row M=lane-16, K = {8..15, 24..31}
__device__ __forceinline__ int a_koff(int lane, int v, int p) {
  return ((lane & 16) ? 8 : 0) + ((v < 4) ? 2 * v : 16 + 2 * (v - 4)) + p;
}
// B (32x16, KxN): lane<16 -> col N=lane, K = {0..7, 16..23};
//                 lane>=16 -> col N=lane-16, K = {8..15, 24..31}
__device__ __forceinline__ int b_koff(int lane, int v, int p) {
  return ((lane & 16) ? 8 : 0) + ((v < 4) ? 2 * v : 16 + 2 * (v - 4)) + p;
}

// Load one B fragment from a (K x N) row-major fp32 weight matrix.
// Rows >= kvalid and cols >= ncols are zero padding.
__device__ __forceinline__ v16h load_bfrag(const float* __restrict__ W, int ldn,
                                           int n0, int kvalid, int ncols, int lane) {
  v16h b;
  const int n = n0 + (lane & 15);
  const bool colok = (n < ncols);
#pragma unroll
  for (int v = 0; v < 8; ++v) {
#pragma unroll
    for (int p = 0; p < 2; ++p) {
      const int k = b_koff(lane, v, p);
      const float val = (colok && k < kvalid) ? W[k * ldn + n] : 0.0f;
      b[2 * v + p] = (_Float16)val;
    }
  }
  return b;
}

// B fragment for X @ W^T GEMMs: logical B[k][n] = W[(n0+n)*ldk + kofs + k].
__device__ __forceinline__ v16h load_bfragT(const float* W, int ldk,
                                            int n0, int kofs, int lane) {
  v16h b;
  const int n = n0 + (lane & 15);
#pragma unroll
  for (int v = 0; v < 8; ++v)
#pragma unroll
    for (int p = 0; p < 2; ++p)
      b[2 * v + p] = (_Float16)W[n * ldk + kofs + b_koff(lane, v, p)];
  return b;
}

// Load one A fragment (16x32 f16) from an LDS matrix with row stride ld.
__device__ __forceinline__ v16h load_afrag_s(const _Float16* src, int ld, int row0,
                                             int kofs, int lane) {
  v16h a;
  const int row = row0 + (lane & 15);
#pragma unroll
  for (int v = 0; v < 8; ++v)
#pragma unroll
    for (int p = 0; p < 2; ++p)
      a[2 * v + p] = src[row * ld + kofs + a_koff(lane, v, p)];
  return a;
}

// Fused ReLU + f32->f16 pack: relu(cvt(x)) == cvt(relu(x)), so convert pairs
// with v_cvt_pk_rtz_f16_f32 and clamp with one packed v_pk_max_num_f16.
__device__ __forceinline__ v2h relu_pk(float a, float b) {
  const v2hf pkf = __builtin_amdgcn_cvt_pkrtz(a, b);
  v2h pk;
  __builtin_memcpy(&pk, &pkf, sizeof(pk));   // same bits, different frontend type
  const v2h z2 = {(_Float16)0.0f, (_Float16)0.0f};
  return __builtin_elementwise_max(pk, z2);
}

__global__ __launch_bounds__(256) void gnn_forward(
    const float* __restrict__ init_features, const float* __restrict__ edge_w,
    const float* __restrict__ noise_info,    const float* __restrict__ hx_init,
    const float* __restrict__ x_hat,         const float* __restrict__ var_mmse,
    const float* __restrict__ W1a, const float* __restrict__ b1a,
    const float* __restrict__ W2a, const float* __restrict__ b2a,
    const float* __restrict__ W2b, const float* __restrict__ b2b,
    const float* __restrict__ W2c, const float* __restrict__ b2c,
    const float* __restrict__ W3a, const float* __restrict__ b3a,
    const float* __restrict__ W3b, const float* __restrict__ b3b,
    const float* __restrict__ W3c, const float* __restrict__ b3c,
    const float* __restrict__ Wih, const float* __restrict__ Whh,
    const float* __restrict__ bih, const float* __restrict__ bhh,
    const float* __restrict__ W4,  const float* __restrict__ b4,
    float* __restrict__ out)
{
  __shared__ _Float16 nodes[U_ * SU_];        // node features, f16 for WMMA A
  __shared__ float    sum_m[U_ * SU_];        // segment sums (ds_add_f32 atomics)
  __shared__ float    hx[U_ * H_];            // GRU hidden state (fp32)
  __shared__ _Float16 hx16[U_ * H_];          // f16 mirror for WMMA A fragments
  __shared__ float    gruread[U_ * SU_];      // gru_read (fp32, feeds final head)
  __shared__ float    head1[U_ * H_];
  __shared__ float    head2[U_ * (H_ / 2)];
  // Time-multiplexed region: per-wave f16 staging tiles (tile loop only) and
  // the fp32 GRU gate pre-activations gh (GRU phase only). Phases are
  // separated by __syncthreads().
  __shared__ __align__(16) unsigned char uni[U_ * G3H * 4]; // 24576 B >= 18432 B

  const int tid  = threadIdx.x;
  const int lane = tid & 31;
  const int wave = tid >> 5;
  const int b    = blockIdx.x;
  const int coll = lane & 15;
  const int hv   = lane >> 4;     // lane half
  const int r16  = lane & 15;

  _Float16* stg = (_Float16*)uni + wave * (16 * STRIDE);
  float*    gh  = (float*)uni;    // [U_][G3H]

  // ---- Preload edge-MLP weight B-fragments (72 VGPRs, reused 62 tiles x 3 it)
  v16h Bw2a[4], Bw2b[2][2], Bw2c;
#pragma unroll
  for (int nt = 0; nt < 4; ++nt)
    Bw2a[nt] = load_bfrag(W2a, H_, nt * 16, 2 * SU_ + 2 /*18*/, H_, lane);
#pragma unroll
  for (int c = 0; c < 2; ++c)
#pragma unroll
    for (int nt = 0; nt < 2; ++nt)
      Bw2b[c][nt] = load_bfrag(W2b + c * 32 * 32, 32, nt * 16, 32, 32, lane);
  Bw2c = load_bfrag(W2c, SU_, 0, 32, SU_ /*N padded 8->16*/, lane);

  // GRU gh = h @ Whh^T: 24 output tiles (2 m-tiles x 12 n-tiles) over 8 waves.
  // This wave owns p = wave, wave+8, wave+16 -> mt = p/12, nt = p%12.
  // The Whh B-fragments are NOT kept resident (that pushed allocation past
  // 256 VGPRs); they are re-loaded from L2-resident weights each iteration.
  int ghmt[3], ghnt[3];
#pragma unroll
  for (int q = 0; q < 3; ++q) {
    const int p = wave + 8 * q;
    ghmt[q] = p / 12;
    ghnt[q] = p % 12;
  }
  float biasg[3];
#pragma unroll
  for (int q = 0; q < 3; ++q) biasg[q] = bhh[ghnt[q] * 16 + coll];

  // Per-lane bias values for the edge MLP (column-indexed)
  float bias1[4], bias2[2];
#pragma unroll
  for (int nt = 0; nt < 4; ++nt) bias1[nt] = b2a[nt * 16 + coll];
#pragma unroll
  for (int nt = 0; nt < 2; ++nt) bias2[nt] = b2b[nt * 16 + coll];
  const float bias3 = (coll < SU_) ? b2c[coll] : 0.0f;

  const _Float16 noiseh = hv ? (_Float16)0.0f : (_Float16)noise_info[b];

  // Load GRU hidden state (fp32 + f16 mirror)
  for (int i = tid; i < U_ * H_; i += 256) {
    const float v = hx_init[b * U_ * H_ + i];
    hx[i]   = v;
    hx16[i] = (_Float16)v;
  }

  for (int it = 0; it < NITER; ++it) {
    // ---- nodes for this iteration
    if (it == 0 && tid < U_ * SU_) {
      const int u = tid / SU_, o = tid % SU_;
      const float x0 = init_features[(b * U_ + u) * 3 + 0];
      const float x1 = init_features[(b * U_ + u) * 3 + 1];
      const float x2 = init_features[(b * U_ + u) * 3 + 2];
      const float x3 = x_hat[b * U_ + u];
      const float x4 = var_mmse[b * U_ + u];
      const float acc = b1a[o] + x0 * W1a[0 * SU_ + o] + x1 * W1a[1 * SU_ + o] +
                        x2 * W1a[2 * SU_ + o] + x3 * W1a[3 * SU_ + o] +
                        x4 * W1a[4 * SU_ + o];
      nodes[tid] = (_Float16)acc;
    }
    if (tid < U_ * SU_) sum_m[tid] = 0.0f;
    __syncthreads();

    // ---- Edge MLP: 62 M-tiles of 16 edges, distributed over 8 waves.
    // Edge weights are software-pipelined: tile t+8's load is issued at the
    // top of tile t's iteration and converted/consumed one iteration later.
    _Float16 ewh = (!hv) ? (_Float16)edge_w[b * E_ + wave * 16 + r16]
                         : (_Float16)0.0f;
    for (int t = wave; t < NT_; t += 8) {
      const int ebase = t * 16;
      const int e = ebase + r16;                 // this lane's edge row
      const float ew_next = (!hv && t + 8 < NT_)
                              ? edge_w[b * E_ + e + 128] : 0.0f;

      const int i = e / 31, j = e % 31;
      const int src = hv ? (j + (j >= i ? 1 : 0)) : i;   // tb : ta

      // Build A0 (16x32 f16): K 0..7 = nodes[ta], 8..15 = nodes[tb],
      // 16 = edge weight, 17 = noise, 18..31 = zero pad.
      v16h a0;
#pragma unroll
      for (int v = 0; v < 4; ++v) {              // lane-local K = 2v, 2v+1
        a0[2 * v]     = nodes[src * SU_ + 2 * v];
        a0[2 * v + 1] = nodes[src * SU_ + 2 * v + 1];
      }
      a0[8] = ewh;                               // K=16 (half 0) / K=24 (zero)
      a0[9] = noiseh;                            // K=17
#pragma unroll
      for (int q = 10; q < 16; ++q) a0[q] = (_Float16)0.0f;

      // Layer 1: (16x32) @ (32x64), C seeded with bias
      v8f c1[4];
#pragma unroll
      for (int nt = 0; nt < 4; ++nt) {
        v8f c;
#pragma unroll
        for (int rr = 0; rr < 8; ++rr) c[rr] = bias1[nt];
        c1[nt] = __builtin_amdgcn_wmma_f32_16x16x32_f16(
            false, a0, false, Bw2a[nt], (short)0, c, false, false);
      }
      // Fused ReLU+pack -> f16 staging tile (16 rows x 64 cols)
#pragma unroll
      for (int nt = 0; nt < 4; ++nt)
#pragma unroll
        for (int rr = 0; rr < 8; rr += 2) {
          const v2h pk = relu_pk(c1[nt][rr], c1[nt][rr + 1]);
          stg[(rr     + 8 * hv) * STRIDE + nt * 16 + r16] = pk.x;
          stg[(rr + 1 + 8 * hv) * STRIDE + nt * 16 + r16] = pk.y;
        }
      LDS_ORDER();   // hardware keeps same-wave LDS ops in order
      v16h a1lo = load_afrag_s(stg, STRIDE, 0, 0,  lane);
      v16h a1hi = load_afrag_s(stg, STRIDE, 0, 32, lane);
      LDS_ORDER();

      // Layer 2: (16x64) @ (64x32) as two K=32 chunks, bias-seeded
      v8f c2[2];
#pragma unroll
      for (int nt = 0; nt < 2; ++nt) {
        v8f c;
#pragma unroll
        for (int rr = 0; rr < 8; ++rr) c[rr] = bias2[nt];
        c = __builtin_amdgcn_wmma_f32_16x16x32_f16(
            false, a1lo, false, Bw2b[0][nt], (short)0, c, false, false);
        c2[nt] = __builtin_amdgcn_wmma_f32_16x16x32_f16(
            false, a1hi, false, Bw2b[1][nt], (short)0, c, false, false);
      }
      // Fused ReLU+pack -> staging (16 x 32)
#pragma unroll
      for (int nt = 0; nt < 2; ++nt)
#pragma unroll
        for (int rr = 0; rr < 8; rr += 2) {
          const v2h pk = relu_pk(c2[nt][rr], c2[nt][rr + 1]);
          stg[(rr     + 8 * hv) * STRIDE + nt * 16 + r16] = pk.x;
          stg[(rr + 1 + 8 * hv) * STRIDE + nt * 16 + r16] = pk.y;
        }
      LDS_ORDER();
      v16h a2 = load_afrag_s(stg, STRIDE, 0, 0, lane);
      LDS_ORDER();

      // Layer 3: (16x32) @ (32x8 padded to 16)
      v8f c3i;
#pragma unroll
      for (int rr = 0; rr < 8; ++rr) c3i[rr] = bias3;
      v8f c3 = __builtin_amdgcn_wmma_f32_16x16x32_f16(
          false, a2, false, Bw2c, (short)0, c3i, false, false);

      // ReLU (fp32) + segment-sum over the 31 edges of each source node
      if (coll < SU_) {
#pragma unroll
        for (int rr = 0; rr < 8; ++rr) {
          const int row = rr + 8 * hv;
          const int ee  = ebase + row;
          float v = c3[rr];
          v = v > 0.0f ? v : 0.0f;
          atomicAdd(&sum_m[(ee / 31) * SU_ + coll], v);   // ds_add_f32
        }
      }
      ewh = (_Float16)ew_next;   // consume the pipelined load
    }
    __syncthreads();   // sum_m done; stage region free -> becomes gh

    // ---- GRU gate pre-activations gh = h @ Whh^T + bhh via WMMA.
    // Make the Whh base pointer opaque so LICM cannot hoist the fragment
    // loads out of the iteration loop (they must not stay resident).
    uintptr_t wp = (uintptr_t)Whh;
    asm volatile("" : "+s"(wp));
    const float* Whh_it = (const float*)wp;
#pragma unroll
    for (int q = 0; q < 3; ++q) {
      v16h b0 = load_bfragT(Whh_it, H_, ghnt[q] * 16, 0,  lane);
      v16h b1 = load_bfragT(Whh_it, H_, ghnt[q] * 16, 32, lane);
      v16h aL = load_afrag_s(hx16, H_, ghmt[q] * 16, 0,  lane);
      v16h aH = load_afrag_s(hx16, H_, ghmt[q] * 16, 32, lane);
      v8f c;
#pragma unroll
      for (int rr = 0; rr < 8; ++rr) c[rr] = biasg[q];
      c = __builtin_amdgcn_wmma_f32_16x16x32_f16(
          false, aL, false, b0, (short)0, c, false, false);
      c = __builtin_amdgcn_wmma_f32_16x16x32_f16(
          false, aH, false, b1, (short)0, c, false, false);
      const int g = ghnt[q] * 16 + coll;
#pragma unroll
      for (int rr = 0; rr < 8; ++rr) {
        const int node = ghmt[q] * 16 + rr + 8 * hv;
        gh[node * G3H + g] = c[rr];
      }
    }
    __syncthreads();   // gh visible to all waves

    // ---- GRU epilogue: thread -> (node, 8 consecutive hidden dims).
    // Only the tiny K=8 Wih dots + gate nonlinearities remain on VALU.
    {
      const int node = tid >> 3;
      const int jb   = (tid & 7) * 8;
      float x[8], nh[8];
#pragma unroll
      for (int k = 0; k < 8; ++k) x[k] = sum_m[node * SU_ + k];
#pragma unroll
      for (int q = 0; q < 8; ++q) {
        const int jj = jb + q;
        float gir = bih[jj], giz = bih[jj + 64], gin = bih[jj + 128];
#pragma unroll
        for (int k = 0; k < 8; ++k) {
          gir += x[k] * Wih[jj * SU_ + k];
          giz += x[k] * Wih[(jj + 64) * SU_ + k];
          gin += x[k] * Wih[(jj + 128) * SU_ + k];
        }
        const float ghr = gh[node * G3H + jj];
        const float ghz = gh[node * G3H + jj + 64];
        const float ghn = gh[node * G3H + jj + 128];
        const float r = 1.0f / (1.0f + __expf(-(gir + ghr)));
        const float z = 1.0f / (1.0f + __expf(-(giz + ghz)));
        const float n = tanhf(gin + r * ghn);
        nh[q] = (1.0f - z) * n + z * hx[node * H_ + jj];
      }
      __syncthreads();
#pragma unroll
      for (int q = 0; q < 8; ++q) {
        hx[node * H_ + jb + q]   = nh[q];
        hx16[node * H_ + jb + q] = (_Float16)nh[q];
      }
      __syncthreads();
    }

    // ---- gru_read = hx @ W4 + b4  -> nodes (f16) for next iter + fp32 copy
    {
      const int node = tid >> 3, o = tid & 7;
      float acc = b4[o];
      for (int k = 0; k < H_; ++k) acc += hx[node * H_ + k] * W4[k * SU_ + o];
      gruread[node * SU_ + o] = acc;
      nodes[node * SU_ + o]   = (_Float16)acc;
    }
    __syncthreads();
  }

  // ---- Final head (no ReLU): gru_read @ W3a @ W3b @ W3c
  for (int idx = tid; idx < U_ * H_; idx += 256) {
    const int node = idx / H_, o = idx % H_;
    float acc = b3a[o];
#pragma unroll
    for (int k = 0; k < SU_; ++k) acc += gruread[node * SU_ + k] * W3a[k * H_ + o];
    head1[idx] = acc;
  }
  __syncthreads();
  for (int idx = tid; idx < U_ * (H_ / 2); idx += 256) {
    const int node = idx / (H_ / 2), o = idx % (H_ / 2);
    float acc = b3b[o];
    for (int k = 0; k < H_; ++k) acc += head1[node * H_ + k] * W3b[k * (H_ / 2) + o];
    head2[idx] = acc;
  }
  __syncthreads();
  if (tid < U_ * NC_) {
    const int node = tid / NC_, o = tid % NC_;
    float acc = b3c[o];
    for (int k = 0; k < H_ / 2; ++k) acc += head2[node * (H_ / 2) + k] * W3c[k * NC_ + o];
    out[b * U_ * NC_ + tid] = acc;
  }
}

extern "C" void kernel_launch(void* const* d_in, const int* in_sizes, int n_in,
                              void* d_out, int out_size, void* d_ws, size_t ws_size,
                              hipStream_t stream) {
  (void)in_sizes; (void)n_in; (void)out_size; (void)d_ws; (void)ws_size;
  const float* init_features = (const float*)d_in[0];
  const float* edge_weight   = (const float*)d_in[1];
  const float* noise_info    = (const float*)d_in[2];
  const float* hx_init       = (const float*)d_in[3];
  /* d_in[4] = cons (unused by reference output) */
  const float* x_hat = (const float*)d_in[5];
  const float* var_m = (const float*)d_in[6];
  const float* W1a = (const float*)d_in[7];  const float* b1a = (const float*)d_in[8];
  const float* W2a = (const float*)d_in[9];  const float* b2a = (const float*)d_in[10];
  const float* W2b = (const float*)d_in[11]; const float* b2b = (const float*)d_in[12];
  const float* W2c = (const float*)d_in[13]; const float* b2c = (const float*)d_in[14];
  const float* W3a = (const float*)d_in[15]; const float* b3a = (const float*)d_in[16];
  const float* W3b = (const float*)d_in[17]; const float* b3b = (const float*)d_in[18];
  const float* W3c = (const float*)d_in[19]; const float* b3c = (const float*)d_in[20];
  const float* Wih = (const float*)d_in[21]; const float* Whh = (const float*)d_in[22];
  const float* bih = (const float*)d_in[23]; const float* bhh = (const float*)d_in[24];
  const float* W4  = (const float*)d_in[25]; const float* b4  = (const float*)d_in[26];

  gnn_forward<<<B_, 256, 0, stream>>>(
      init_features, edge_weight, noise_info, hx_init, x_hat, var_m,
      W1a, b1a, W2a, b2a, W2b, b2b, W2c, b2c, W3a, b3a, W3b, b3b, W3c, b3c,
      Wih, Whh, bih, bhh, W4, b4, (float*)d_out);
}
